// Spatial_Attention_layer_48644799595034
// MI455X (gfx1250) — compile-verified, matrix operand order checked by reading, not run on hardware
//
#include <hip/hip_runtime.h>
#include <hip/hip_bf16.h>
#include <math.h>

// Sizes fixed by the reference
#define Bd 64
#define Nd 1024
#define Cd 64

typedef __attribute__((ext_vector_type(16))) _Float16 v16h;
typedef __attribute__((ext_vector_type(8)))  _Float16 v8h;
typedef __attribute__((ext_vector_type(8)))  float    v8f;

// ---------------------------------------------------------------------------
// 16-bit A-fragment (16x32 MxK) loader per CDNA5 ISA layout:
//   lane L: m = L&15 ; L<16 holds K {0..7,16..23}, L>=16 holds K {8..15,24..31}
// B-fragment (32x16 KxN) mirrors A when B is stored column-major (NxK
// row-major), so the same loader serves both operands.
// ---------------------------------------------------------------------------
__device__ __forceinline__ v16h load_frag16(const _Float16* __restrict__ base,
                                            int row0, int k0, int ld) {
    const int lane = threadIdx.x & 31;
    const int m    = lane & 15;
    const int sel  = lane >> 4;                    // 0 or 1
    const _Float16* p = base + (size_t)(row0 + m) * ld + k0 + sel * 8;
    union { v16h v; v8h h[2]; } u;
    u.h[0] = *(const v8h*)(p);                     // K sel*8 .. sel*8+7
    u.h[1] = *(const v8h*)(p + 16);                // K sel*8+16 .. sel*8+23
    return u.v;
}

__device__ __forceinline__ v8f wmma_f16(v16h a, v16h b, v8f c) {
    return __builtin_amdgcn_wmma_f32_16x16x32_f16(
        /*neg_a=*/false, a, /*neg_b=*/false, b,
        /*c_mod=*/(short)0, c, /*reuse_a=*/false, /*reuse_b=*/false);
}

// native v_rcp_f32 (single TRANS op, no div_scale/fma fixup chain)
__device__ __forceinline__ float fast_rcp(float x) {
    return __builtin_amdgcn_rcpf(x);
}

// ---------------------------------------------------------------------------
// Pass 0: f32 -> f16 convert (grid-stride)
// ---------------------------------------------------------------------------
__global__ void __launch_bounds__(256)
k_cvt_f16(const float* __restrict__ src, _Float16* __restrict__ dst, int n) {
    for (int i = blockIdx.x * blockDim.x + threadIdx.x; i < n;
         i += gridDim.x * blockDim.x)
        dst[i] = (_Float16)src[i];
}

// ---------------------------------------------------------------------------
// Pass 1: score^T[b][k][j] = sigmoid( gram[b][j][k]*diag(W2)[k] + b_s[j][k] )
// One wave per 16x16 tile. gram tile via 2x WMMA over K=C=64.
// D-fragment: lane L holds col n=L&15, rows (L>=16?8:0)+r for r=0..7 -> the
// transposed f16 store is one contiguous 16B store per lane.
// ---------------------------------------------------------------------------
__global__ void __launch_bounds__(256)
k_score(const _Float16* __restrict__ xh, const float* __restrict__ W2,
        const float* __restrict__ bs, _Float16* __restrict__ scoreT) {
    const int w  = blockIdx.x * 8 + (threadIdx.x >> 5);   // global wave id
    const int kt = w & 63;
    const int jt = (w >> 6) & 63;
    const int b  = w >> 12;

    const _Float16* xb = xh + (size_t)b * Nd * Cd;
    v16h a0 = load_frag16(xb, jt * 16, 0,  Cd);   // rows j, K = c
    v16h a1 = load_frag16(xb, jt * 16, 32, Cd);
    v16h b0 = load_frag16(xb, kt * 16, 0,  Cd);   // B col-major == x row-major
    v16h b1 = load_frag16(xb, kt * 16, 32, Cd);

    v8f acc = {0.f, 0.f, 0.f, 0.f, 0.f, 0.f, 0.f, 0.f};
    acc = wmma_f16(a0, b0, acc);
    acc = wmma_f16(a1, b1, acc);

    const int lane = threadIdx.x & 31;
    const int n    = lane & 15;
    const int off  = (lane >> 4) * 8;
    const int kcol = kt * 16 + n;
    const int jbase = jt * 16 + off;
    const float dval = W2[(size_t)kcol * (Nd + 1)];   // diagonal of W_2

    v8h o;
#pragma unroll
    for (int r = 0; r < 8; ++r) {
        float s   = acc[r] * dval + bs[(size_t)(jbase + r) * Nd + kcol];
        // sigmoid via native v_exp_f32 + v_rcp_f32 (feeds an f16 store)
        float sig = fast_rcp(1.0f + __expf(-s));
        o[r] = (_Float16)sig;
    }
    *(v8h*)(scoreT + (size_t)b * Nd * Nd + (size_t)kcol * Nd + jbase) = o;
}

// ---------------------------------------------------------------------------
// Pass 2 (fused): Z[:, kstripe] = V_s @ score[b][:, kstripe]  -> LDS (64 KB)
// then softmax over rows (axis=1) entirely in LDS, coalesced global write.
// Block = 256 threads (8 waves); wave w owns 8 row-tiles (128 rows).
// Grid = B * (N/16) = 4096.
// ---------------------------------------------------------------------------
__global__ void __launch_bounds__(256)
k_gemm_softmax(const _Float16* __restrict__ vsh,
               const _Float16* __restrict__ scoreT,
               float* __restrict__ out) {
    __shared__ float Zl[Nd * 16];                 // 1024 rows x 16 cols = 64 KB

    const int b  = blockIdx.x >> 6;
    const int kt = blockIdx.x & 63;
    const _Float16* sb = scoreT + (size_t)b * Nd * Nd;

    const int wave = threadIdx.x >> 5;
    const int lane = threadIdx.x & 31;

    v8f acc[8];
#pragma unroll
    for (int t = 0; t < 8; ++t) acc[t] = (v8f){0.f,0.f,0.f,0.f,0.f,0.f,0.f,0.f};

    for (int k0 = 0; k0 < Nd; k0 += 32) {
        // hide L2 latency for the V_s stream of the next K-slice
        // (lowers to global_prefetch_b8: no counter, no VGPR return)
        if (k0 + 32 < Nd) {
            const _Float16* pf =
                vsh + (size_t)(wave * 8 * 16 + (lane & 15)) * Nd + (k0 + 32);
            __builtin_prefetch(pf, 0, 3);
        }
        v16h bf = load_frag16(sb, kt * 16, k0, Nd);   // scoreT rows = out cols
#pragma unroll
        for (int t = 0; t < 8; ++t) {
            v16h af = load_frag16(vsh, (wave * 8 + t) * 16, k0, Nd);
            acc[t] = wmma_f16(af, bf, acc[t]);        // reuses bf 8x
        }
    }

    // spill accumulators: Zl[i*16 + n]
    const int n   = lane & 15;
    const int off = (lane >> 4) * 8;
#pragma unroll
    for (int t = 0; t < 8; ++t) {
        const int i0 = (wave * 8 + t) * 16 + off;
#pragma unroll
        for (int r = 0; r < 8; ++r) Zl[(i0 + r) * 16 + n] = acc[t][r];
    }
    __syncthreads();

    // softmax over the 1024 rows of each of the 16 columns.
    // col = tid>>4 keeps the 16 cooperating threads inside one half-wave so
    // the cross-thread reductions are pure __shfl_xor (no LDS scratch).
    const int col = threadIdx.x >> 4;   // 0..15
    const int grp = threadIdx.x & 15;   // 0..15 (row phase)

    float m = -INFINITY;
#pragma unroll 4
    for (int s = 0; s < 64; ++s)
        m = fmaxf(m, Zl[(grp + 16 * s) * 16 + col]);
#pragma unroll
    for (int d = 1; d < 16; d <<= 1) m = fmaxf(m, __shfl_xor(m, d, 32));

    float p = 0.f;
#pragma unroll 4
    for (int s = 0; s < 64; ++s) {
        const int idx = (grp + 16 * s) * 16 + col;
        float e = __expf(Zl[idx] - m);
        Zl[idx] = e;                    // thread-exclusive rows: no hazard
        p += e;
    }
#pragma unroll
    for (int d = 1; d < 16; d <<= 1) p += __shfl_xor(p, d, 32);
    const float inv = fast_rcp(p);      // 1/sum via native v_rcp_f32
#pragma unroll 4
    for (int s = 0; s < 64; ++s)
        Zl[(grp + 16 * s) * 16 + col] *= inv;
    __syncthreads();

    // coalesced write-out: consecutive tids -> consecutive columns (64B rows)
    float* ob = out + (size_t)b * Nd * Nd + kt * 16;
    const int c2 = threadIdx.x & 15;
    const int r2 = threadIdx.x >> 4;
#pragma unroll 4
    for (int s = 0; s < 64; ++s) {
        const int i = r2 + 16 * s;
        ob[(size_t)i * Nd + c2] = Zl[i * 16 + c2];
    }
}

// ---------------------------------------------------------------------------
// Launch
// ---------------------------------------------------------------------------
extern "C" void kernel_launch(void* const* d_in, const int* in_sizes, int n_in,
                              void* d_out, int out_size, void* d_ws, size_t ws_size,
                              hipStream_t stream) {
    const float* x  = (const float*)d_in[0];   // (B,N,C)
    const float* W2 = (const float*)d_in[1];   // (N,N)
    const float* bs = (const float*)d_in[2];   // (1,N,N)
    const float* Vs = (const float*)d_in[3];   // (N,N)
    float* out = (float*)d_out;                // (B,N,N)

    // Workspace layout (bytes):
    //   [0, 8MB)      x_f16      B*N*C * 2
    //   [8MB, 10MB)   Vs_f16     N*N   * 2
    //   [16MB, 144MB) score^T    B*N*N * 2   (f16, transposed per batch)
    char* ws = (char*)d_ws;
    _Float16* xh     = (_Float16*)(ws);
    _Float16* vsh    = (_Float16*)(ws + (size_t)8  * 1024 * 1024);
    _Float16* scoreT = (_Float16*)(ws + (size_t)16 * 1024 * 1024);

    const int nx  = Bd * Nd * Cd;   // 4,194,304
    const int nvs = Nd * Nd;        // 1,048,576

    k_cvt_f16<<<nx  / 256, 256, 0, stream>>>(x,  xh,  nx);
    k_cvt_f16<<<nvs / 256, 256, 0, stream>>>(Vs, vsh, nvs);

    // one wave per 16x16 tile: B * 64 * 64 waves / 8 waves per block
    k_score<<<(Bd * 64 * 64) / 8, 256, 0, stream>>>(xh, W2, bs, scoreT);

    // one block per (batch, 16-column stripe)
    k_gemm_softmax<<<Bd * (Nd / 16), 256, 0, stream>>>(vsh, scoreT, out);
}
